// Qnet_49392123904652
// MI455X (gfx1250) — matrix operand order repaired and checked by reading, not compiled, not run on hardware
//
#include <hip/hip_runtime.h>
#include <hip/hip_bf16.h>
#include <math.h>
#include <stdint.h>

typedef __bf16 bf16_t;
typedef __attribute__((ext_vector_type(16))) __bf16 v16bf;
typedef __attribute__((ext_vector_type(8)))  __bf16 v8bf;
typedef __attribute__((ext_vector_type(8)))  float  v8f;
typedef __attribute__((ext_vector_type(4)))  int    v4i;
typedef __attribute__((address_space(1))) v4i gv4i;   // global int4
typedef __attribute__((address_space(3))) v4i lv4i;   // LDS int4

#define SEQ   16
#define BSZ   256
#define NIN   5040      // INPUT_SIZE
#define NINP  5056      // K padded to 32
#define HID   1260
#define HIDP  1280
#define G4    5040      // 4*HID
#define ROWS  4096      // SEQ*BSZ
#define HW    126       // 9*14
#define CONVA_N 1764    // 14*126
#define CONVB_N 1512    // 12*126
#define CONVB_NP 1536
#define HEADH 630
#define HEADHP 640
#define NHEADS 20
#define LINNP 1264
#define KPAD  40        // LDS row pitch (32 K elems + 8 pad) to spread banks

// ----------------------------------------------------- async LDS staging path
#if defined(__has_builtin)
# if __has_builtin(__builtin_amdgcn_global_load_async_to_lds_b128)
#  define HAVE_ASYNC_LDS 1
# else
#  define HAVE_ASYNC_LDS 0
# endif
#else
# define HAVE_ASYNC_LDS 0
#endif

__device__ __forceinline__ void cp16(const bf16_t* g, bf16_t* l) {
#if HAVE_ASYNC_LDS
    // AS1 address == flat address for global; AS3 address == low 32 bits of
    // the flat LDS-aperture address.
    gv4i* gp = (gv4i*)(unsigned long long)(uintptr_t)(const void*)g;
    lv4i* lp = (lv4i*)(unsigned)(uintptr_t)(void*)l;
    __builtin_amdgcn_global_load_async_to_lds_b128(gp, lp, 0, 0);
#else
    *(v8bf*)l = *(const v8bf*)g;
#endif
}
__device__ __forceinline__ void wait_async0() {
#if HAVE_ASYNC_LDS
# if __has_builtin(__builtin_amdgcn_s_wait_asynccnt)
    __builtin_amdgcn_s_wait_asynccnt(0);
# else
    asm volatile("s_wait_asynccnt 0x0" ::: "memory");
# endif
#endif
}
__device__ __forceinline__ void wait_async3() {
#if HAVE_ASYNC_LDS
# if __has_builtin(__builtin_amdgcn_s_wait_asynccnt)
    __builtin_amdgcn_s_wait_asynccnt(3);
# else
    asm volatile("s_wait_asynccnt 0x3" ::: "memory");
# endif
#endif
}

// ---------------------------------------------------------------- fragments
__device__ __forceinline__ v16bf ld_fragA(const bf16_t* q) {
    // lane covers A[m][hf*8 + 0..7] and A[m][hf*8 + 16..23]
    v8bf lo = *(const v8bf*)q;
    v8bf hi = *(const v8bf*)(q + 16);
    v16bf f;
#pragma unroll
    for (int i = 0; i < 8; ++i) { f[i] = lo[i]; f[8 + i] = hi[i]; }
    return f;
}
__device__ __forceinline__ v16bf ld_fragB(const bf16_t* q) {
    // lane covers B-row (W[n]) K = hf*8 + 0..7 and hf*8 + 16..23
    v8bf lo = *(const v8bf*)q;
    v8bf hi = *(const v8bf*)(q + 16);
    v16bf f;
#pragma unroll
    for (int i = 0; i < 8; ++i) { f[i] = lo[i]; f[8 + i] = hi[i]; }
    return f;
}
__device__ __forceinline__ v8f wmma_bf16(v16bf a, v16bf b, v8f c) {
    return __builtin_amdgcn_wmma_f32_16x16x32_bf16(false, a, false, b,
                                                   (short)0, c, false, false);
}

// ------------------------------------------------------------------- GEMM
// C[M,N](f32,ldc) = A[M,K](bf16,lda) * B[N,K]^T(bf16,ldb)
// K%32==0, M%64==0. Block 256 thr = 8 waves -> 64x128 C tile.
// Double-buffered LDS staging (async-to-LDS when available); each wave owns a
// 16-wide column strip and 4 M-tiles reusing one B fragment (4 wmma / K-step).
__global__ __launch_bounds__(256) void gemm_bf16_kernel(
    const bf16_t* __restrict__ A, const bf16_t* __restrict__ B,
    float* __restrict__ C, int M, int N, int K, int lda, int ldb, int ldc,
    long long bA, long long bB, long long bC) {
    __shared__ bf16_t As[2][64 * KPAD];
    __shared__ bf16_t Bs[2][128 * KPAD];
    const int tid = threadIdx.x;
    const int wave = tid >> 5, lane = tid & 31;
    const int r = lane & 15, hf = lane >> 4;
    const int m0b = blockIdx.y * 64;
    const int n0b = blockIdx.x * 128;
    const bf16_t* a = A + (long long)blockIdx.z * bA;
    const bf16_t* b = B + (long long)blockIdx.z * bB;
    float* c = C + (long long)blockIdx.z * bC;

    // cooperative staging: A 64x32 (1 x 16B per thread), B 128x32 (2 x 16B)
    const int arow = tid >> 2, achk = (tid & 3) * 8;
    const long long aoff = (long long)(m0b + arow) * lda + achk;
    const int br0 = tid >> 2, br1 = (tid + 256) >> 2, bchk = (tid & 3) * 8;
    int bg0 = n0b + br0; if (bg0 >= N) bg0 = N - 1;   // clamp: safe reads only
    int bg1 = n0b + br1; if (bg1 >= N) bg1 = N - 1;
    const long long boff0 = (long long)bg0 * ldb + bchk;
    const long long boff1 = (long long)bg1 * ldb + bchk;

    v8f acc[4] = {{}, {}, {}, {}};

    // prologue
    cp16(a + aoff, &As[0][arow * KPAD + achk]);
    cp16(b + boff0, &Bs[0][br0 * KPAD + bchk]);
    cp16(b + boff1, &Bs[0][br1 * KPAD + bchk]);

    for (int k = 0; k < K; k += 32) {
        const int cur = (k >> 5) & 1;
        if (k + 32 < K) {
            const int nxt = cur ^ 1;
#if !HAVE_ASYNC_LDS
            __builtin_prefetch(a + aoff + k + 64, 0, 0);
#endif
            cp16(a + aoff + (k + 32), &As[nxt][arow * KPAD + achk]);
            cp16(b + boff0 + (k + 32), &Bs[nxt][br0 * KPAD + bchk]);
            cp16(b + boff1 + (k + 32), &Bs[nxt][br1 * KPAD + bchk]);
            wait_async3();                 // in-order: older buffer complete
        } else {
            wait_async0();
        }
        __syncthreads();                   // staged data visible block-wide

        const bf16_t* bp = &Bs[cur][(wave * 16 + r) * KPAD + hf * 8];
        v16bf bfr = ld_fragB(bp);
        const bf16_t* ap = &As[cur][r * KPAD + hf * 8];
#pragma unroll
        for (int mt = 0; mt < 4; ++mt)
            acc[mt] = wmma_bf16(ld_fragA(ap + mt * 16 * KPAD), bfr, acc[mt]);
        __syncthreads();                   // done reading before overwrite
    }

    const int col = n0b + wave * 16 + r;
    if (col < N) {
#pragma unroll
        for (int mt = 0; mt < 4; ++mt) {
            float* cp = c + (size_t)(m0b + mt * 16 + hf * 8) * ldc + col;
#pragma unroll
            for (int i = 0; i < 8; ++i) cp[(size_t)i * ldc] = acc[mt][i];
        }
    }
}

// Final heads: out[20,256,16] = hxbf[k][256,640] * headO[k][16,640]^T + bias
__global__ __launch_bounds__(256) void gemm_headO_kernel(
    const bf16_t* __restrict__ hxbf, const bf16_t* __restrict__ Wb,
    const float* __restrict__ bias, float* __restrict__ out) {
    int wave = threadIdx.x >> 5, lane = threadIdx.x & 31;
    int head = blockIdx.z;
    int m0 = (blockIdx.x * 8 + wave) * 16;
    const bf16_t* a = hxbf + (size_t)head * BSZ * HEADHP;
    const bf16_t* b = Wb + (size_t)head * 16 * HEADHP;
    int r = lane & 15, hf = lane >> 4;
    const bf16_t* ap = a + (size_t)(m0 + r) * HEADHP + hf * 8;
    const bf16_t* bp = b + (size_t)r * HEADHP + hf * 8;
    v8f acc = {};
#pragma unroll 4
    for (int k = 0; k < HEADHP; k += 32)
        acc = wmma_bf16(ld_fragA(ap + k), ld_fragB(bp + k), acc);
    float bb = bias[head * 16 + r];
    float* cp = out + (size_t)head * BSZ * 16 + (size_t)(m0 + hf * 8) * 16 + r;
#pragma unroll
    for (int i = 0; i < 8; ++i) cp[(size_t)i * 16] = acc[i] + bb;
}

// ------------------------------------------------------- convert/pad to bf16
__global__ __launch_bounds__(256) void cvt_pad_kernel(
    const float* __restrict__ src, bf16_t* __restrict__ dst,
    int R, int C, int Rp, int Cp, int slda, long long sG, long long total) {
    long long i = (long long)blockIdx.x * 256 + threadIdx.x;
    if (i >= total) return;
    int cp = (int)(i % Cp);
    long long t = i / Cp;
    int rp = (int)(t % Rp);
    long long g = t / Rp;
    float v = 0.f;
    if (rp < R && cp < C) v = src[g * sG + (long long)rp * slda + cp];
    dst[i] = (bf16_t)v;
}

// ------------------------------------------------------------- LSTM gates
__device__ __forceinline__ float sigf(float x) { return 1.f / (1.f + expf(-x)); }

__global__ __launch_bounds__(256) void lstm_gates_kernel(
    const float* __restrict__ Zx, const float* __restrict__ Zh,
    const float* __restrict__ bih, const float* __restrict__ bhh,
    float* __restrict__ c, float* __restrict__ h,
    bf16_t* __restrict__ hbf, bf16_t* __restrict__ ysbf, int t) {
    int tid = blockIdx.x * 256 + threadIdx.x;
    if (tid >= BSZ * HID) return;
    int b = tid / HID, j = tid - b * HID;
    const float* zx = Zx + (size_t)(t * BSZ + b) * G4;
    const float* zh = Zh + (size_t)b * G4;
    float zi = zx[j]            + zh[j]            + bih[j]            + bhh[j];
    float zf = zx[HID + j]      + zh[HID + j]      + bih[HID + j]      + bhh[HID + j];
    float zg = zx[2 * HID + j]  + zh[2 * HID + j]  + bih[2 * HID + j]  + bhh[2 * HID + j];
    float zo = zx[3 * HID + j]  + zh[3 * HID + j]  + bih[3 * HID + j]  + bhh[3 * HID + j];
    size_t hi = (size_t)b * HIDP + j;
    float cn = sigf(zf) * c[hi] + sigf(zi) * tanhf(zg);
    float hn = sigf(zo) * tanhf(cn);
    c[hi] = cn; h[hi] = hn;
    bf16_t hb = (bf16_t)hn;
    hbf[hi] = hb;
    if (ysbf) ysbf[(size_t)(t * BSZ + b) * HIDP + j] = hb;
}

// ------------------------------------------------------------------ conv3x3
__global__ __launch_bounds__(256) void conv3x3_kernel(
    const float* __restrict__ in, int inStride, int Cin,
    const float* __restrict__ w, const float* __restrict__ bias,
    float* __restrict__ out, int outStride, int Cout) {
    int tid = blockIdx.x * 256 + threadIdx.x;
    int total = BSZ * Cout * HW;
    if (tid >= total) return;
    int x = tid % 14;
    int y = (tid / 14) % 9;
    int co = (tid / HW) % Cout;
    int b = tid / (HW * Cout);
    float acc = bias[co];
    const float* ib = in + (size_t)b * inStride;
    for (int ci = 0; ci < Cin; ++ci) {
        const float* ic = ib + ci * HW;
        const float* wc = w + ((co * Cin + ci) * 9);
#pragma unroll
        for (int ky = 0; ky < 3; ++ky) {
            int yy = y + ky - 1;
            if (yy < 0 || yy >= 9) continue;
#pragma unroll
            for (int kx = 0; kx < 3; ++kx) {
                int xx = x + kx - 1;
                if (xx < 0 || xx >= 14) continue;
                acc += ic[yy * 14 + xx] * wc[ky * 3 + kx];
            }
        }
    }
    out[(size_t)b * outStride + co * HW + y * 14 + x] = acc;
}

// ----------------------------------------------- k-winners (exact radix select)
__global__ __launch_bounds__(256) void kwinners_kernel(
    float* __restrict__ data, int rowStride, int n, int k,
    const float* __restrict__ duty, int bdiv, int dgRows, int dgStride) {
    int row = blockIdx.x;
    float* x = data + (size_t)row * rowStride;
    const float* d = duty + (size_t)(row / dgRows) * dgStride;
    __shared__ unsigned bins[256];
    __shared__ unsigned s_digit, s_k;
    unsigned prefix = 0, maskhi = 0, kk = (unsigned)k;
    for (int pass = 0; pass < 4; ++pass) {
        int shift = 24 - 8 * pass;
        for (int i = threadIdx.x; i < 256; i += 256) bins[i] = 0;
        __syncthreads();
        for (int j = threadIdx.x; j < n; j += 256) {
            float v = x[j]; v = v > 0.f ? v : 0.f;
            float bst = v * expf(2.0f * (0.33f - d[j / bdiv]));
            unsigned u = __float_as_uint(bst);
            if ((u & maskhi) == prefix) atomicAdd(&bins[(u >> shift) & 255u], 1u);
        }
        __syncthreads();
        if (threadIdx.x == 0) {
            unsigned cum = 0, dsel = 0, rem = kk;
            for (int dd = 255; dd >= 0; --dd) {
                unsigned cnt = bins[dd];
                if (cum + cnt >= kk) { dsel = (unsigned)dd; rem = kk - cum; break; }
                cum += cnt;
            }
            s_digit = dsel; s_k = rem;
        }
        __syncthreads();
        prefix |= (s_digit << shift);
        maskhi |= (255u << shift);
        kk = s_k;
    }
    float thr = __uint_as_float(prefix);
    for (int j = threadIdx.x; j < n; j += 256) {
        float v = x[j]; float rl = v > 0.f ? v : 0.f;
        float bst = rl * expf(2.0f * (0.33f - d[j / bdiv]));
        x[j] = (bst >= thr) ? rl : 0.f;
    }
}

// --------------------------------------------------------------- bias adds
__global__ __launch_bounds__(256) void bias_rows_kernel(
    float* __restrict__ X, const float* __restrict__ b, int R, int C, int ld) {
    int tid = blockIdx.x * 256 + threadIdx.x;
    if (tid >= R * C) return;
    int r = tid / C, j = tid - r * C;
    X[(size_t)r * ld + j] += b[j];
}
__global__ __launch_bounds__(256) void bias_heads_kernel(
    float* __restrict__ hx, const float* __restrict__ b) {
    int tid = blockIdx.x * 256 + threadIdx.x;
    if (tid >= NHEADS * BSZ * HEADH) return;
    int head = tid / (BSZ * HEADH);
    int rem = tid - head * BSZ * HEADH;
    int bb = rem / HEADH, j = rem - bb * HEADH;
    hx[(size_t)(head * BSZ + bb) * HEADHP + j] += b[head * HEADH + j];
}

// --------------------------------------------------------------------- host
static void launch_gemm(const bf16_t* A, const bf16_t* B, float* C,
                        int M, int N, int K, int lda, int ldb, int ldc,
                        long long bA, long long bB, long long bC, int batch,
                        hipStream_t s) {
    dim3 grid((N + 127) / 128, M / 64, batch);
    gemm_bf16_kernel<<<grid, 256, 0, s>>>(A, B, C, M, N, K, lda, ldb, ldc, bA, bB, bC);
}
static void launch_cvt(const float* src, bf16_t* dst, int G, int R, int C,
                       int Rp, int Cp, int slda, hipStream_t s) {
    long long total = (long long)G * Rp * Cp;
    long long sG = (long long)R * slda;
    int blocks = (int)((total + 255) / 256);
    cvt_pad_kernel<<<blocks, 256, 0, s>>>(src, dst, R, C, Rp, Cp, slda, sG, total);
}

extern "C" void kernel_launch(void* const* d_in, const int* in_sizes, int n_in,
                              void* d_out, int out_size, void* d_ws, size_t ws_size,
                              hipStream_t stream) {
    const float* x       = (const float*)d_in[0];
    const float* Wih0    = (const float*)d_in[1];
    const float* Whh0    = (const float*)d_in[2];
    const float* bih0    = (const float*)d_in[3];
    const float* bhh0    = (const float*)d_in[4];
    const float* Wih1    = (const float*)d_in[5];
    const float* Whh1    = (const float*)d_in[6];
    const float* bih1    = (const float*)d_in[7];
    const float* bhh1    = (const float*)d_in[8];
    const float* convA_w = (const float*)d_in[9];
    const float* convA_b = (const float*)d_in[10];
    const float* convB_w = (const float*)d_in[11];
    const float* convB_b = (const float*)d_in[12];
    const float* lin_w   = (const float*)d_in[13];
    const float* lin_b   = (const float*)d_in[14];
    const float* headX_w = (const float*)d_in[15];
    const float* headX_b = (const float*)d_in[16];
    const float* headO_w = (const float*)d_in[17];
    const float* headO_b = (const float*)d_in[18];
    const float* duty_cA = (const float*)d_in[19];
    const float* duty_cB = (const float*)d_in[20];
    const float* duty_x  = (const float*)d_in[21];
    float* out = (float*)d_out;

    char* ws = (char*)d_ws;
    size_t off = 0;
    auto alloc = [&](size_t bytes) -> size_t {
        size_t o = (off + 255) & ~(size_t)255; off = o + bytes; return o;
    };
    size_t o_Xbf   = alloc((size_t)ROWS * NINP * 2);
    size_t o_W0bf  = alloc((size_t)G4 * NINP * 2);
    size_t o_Wh0bf = alloc((size_t)G4 * HIDP * 2);
    size_t o_W1bf  = alloc((size_t)G4 * HIDP * 2);
    size_t o_Wh1bf = alloc((size_t)G4 * HIDP * 2);
    size_t o_Zx    = alloc((size_t)ROWS * G4 * 4);
    size_t o_ys0   = alloc((size_t)ROWS * HIDP * 2);
    size_t o_h     = alloc((size_t)BSZ * HIDP * 4);
    size_t o_c     = alloc((size_t)BSZ * HIDP * 4);
    size_t o_hbf   = alloc((size_t)BSZ * HIDP * 2);
    size_t o_Zh    = alloc((size_t)BSZ * G4 * 4);
    size_t o_cA    = alloc((size_t)BSZ * CONVA_N * 4);
    size_t o_cB    = alloc((size_t)BSZ * CONVB_N * 4);
    size_t o_himg  = alloc((size_t)BSZ * CONVB_NP * 2);
    size_t o_linW  = alloc((size_t)LINNP * CONVB_NP * 2);
    size_t o_h2    = alloc((size_t)BSZ * LINNP * 4);
    size_t o_h2bf  = alloc((size_t)BSZ * HIDP * 2);
    size_t o_hXw   = alloc((size_t)NHEADS * HEADHP * HIDP * 2);
    size_t o_hx    = alloc((size_t)NHEADS * BSZ * HEADHP * 4);
    size_t o_hxbf  = alloc((size_t)NHEADS * BSZ * HEADHP * 2);
    size_t o_hOw   = alloc((size_t)NHEADS * 16 * HEADHP * 2);
    (void)ws_size;

    bf16_t* Xbf   = (bf16_t*)(ws + o_Xbf);
    bf16_t* W0bf  = (bf16_t*)(ws + o_W0bf);
    bf16_t* Wh0bf = (bf16_t*)(ws + o_Wh0bf);
    bf16_t* W1bf  = (bf16_t*)(ws + o_W1bf);
    bf16_t* Wh1bf = (bf16_t*)(ws + o_Wh1bf);
    float*  Zx    = (float*)(ws + o_Zx);
    bf16_t* ys0   = (bf16_t*)(ws + o_ys0);
    float*  h     = (float*)(ws + o_h);
    float*  c     = (float*)(ws + o_c);
    bf16_t* hbf   = (bf16_t*)(ws + o_hbf);
    float*  Zh    = (float*)(ws + o_Zh);
    float*  cA    = (float*)(ws + o_cA);
    float*  cB    = (float*)(ws + o_cB);
    bf16_t* himg  = (bf16_t*)(ws + o_himg);
    bf16_t* linW  = (bf16_t*)(ws + o_linW);
    float*  h2    = (float*)(ws + o_h2);
    bf16_t* h2bf  = (bf16_t*)(ws + o_h2bf);
    bf16_t* hXw   = (bf16_t*)(ws + o_hXw);
    float*  hx    = (float*)(ws + o_hx);
    bf16_t* hxbf  = (bf16_t*)(ws + o_hxbf);
    bf16_t* hOw   = (bf16_t*)(ws + o_hOw);

    // --- weight / input conversion to padded bf16
    launch_cvt(x,       Xbf,   1, ROWS, NIN, ROWS, NINP, NIN, stream);
    launch_cvt(Wih0,    W0bf,  1, G4, NIN, G4, NINP, NIN, stream);
    launch_cvt(Whh0,    Wh0bf, 1, G4, HID, G4, HIDP, HID, stream);
    launch_cvt(Wih1,    W1bf,  1, G4, HID, G4, HIDP, HID, stream);
    launch_cvt(Whh1,    Wh1bf, 1, G4, HID, G4, HIDP, HID, stream);
    launch_cvt(lin_w,   linW,  1, HID, CONVB_N, LINNP, CONVB_NP, CONVB_N, stream);
    launch_cvt(headX_w, hXw,   NHEADS, HEADH, HID, HEADHP, HIDP, HID, stream);
    launch_cvt(headO_w, hOw,   NHEADS, 16, HEADH, 16, HEADHP, HEADH, stream);

    // --- layer 0: hoisted input GEMM, then 16 recurrent steps
    (void)hipMemsetAsync(h,   0, (size_t)BSZ * HIDP * 4, stream);
    (void)hipMemsetAsync(c,   0, (size_t)BSZ * HIDP * 4, stream);
    (void)hipMemsetAsync(hbf, 0, (size_t)BSZ * HIDP * 2, stream);
    (void)hipMemsetAsync(ys0, 0, (size_t)ROWS * HIDP * 2, stream);

    launch_gemm(Xbf, W0bf, Zx, ROWS, G4, NINP, NINP, NINP, G4, 0, 0, 0, 1, stream);
    int gateBlocks = (BSZ * HID + 255) / 256;
    for (int t = 0; t < SEQ; ++t) {
        launch_gemm(hbf, Wh0bf, Zh, BSZ, G4, HIDP, HIDP, HIDP, G4, 0, 0, 0, 1, stream);
        lstm_gates_kernel<<<gateBlocks, 256, 0, stream>>>(Zx, Zh, bih0, bhh0, c, h, hbf, ys0, t);
    }

    // --- layer 1
    (void)hipMemsetAsync(h,   0, (size_t)BSZ * HIDP * 4, stream);
    (void)hipMemsetAsync(c,   0, (size_t)BSZ * HIDP * 4, stream);
    (void)hipMemsetAsync(hbf, 0, (size_t)BSZ * HIDP * 2, stream);
    launch_gemm(ys0, W1bf, Zx, ROWS, G4, HIDP, HIDP, HIDP, G4, 0, 0, 0, 1, stream);
    for (int t = 0; t < SEQ; ++t) {
        launch_gemm(hbf, Wh1bf, Zh, BSZ, G4, HIDP, HIDP, HIDP, G4, 0, 0, 0, 1, stream);
        lstm_gates_kernel<<<gateBlocks, 256, 0, stream>>>(Zx, Zh, bih1, bhh1, c, h, hbf, (bf16_t*)nullptr, t);
    }
    // h (f32, ld HIDP) == latent [256,1260]

    // --- conv A + kwinners2d
    conv3x3_kernel<<<(BSZ * 14 * HW + 255) / 256, 256, 0, stream>>>(
        h, HIDP, 10, convA_w, convA_b, cA, CONVA_N, 14);
    kwinners_kernel<<<BSZ, 256, 0, stream>>>(cA, CONVA_N, CONVA_N, 582, duty_cA, HW, 1 << 20, 0);
    // --- conv B + kwinners2d
    conv3x3_kernel<<<(BSZ * 12 * HW + 255) / 256, 256, 0, stream>>>(
        cA, CONVA_N, 14, convB_w, convB_b, cB, CONVB_N, 12);
    kwinners_kernel<<<BSZ, 256, 0, stream>>>(cB, CONVB_N, CONVB_N, 498, duty_cB, HW, 1 << 20, 0);

    // --- linear: h2[256,1260] = cB[256,1512] @ lin_w^T + lin_b
    launch_cvt(cB, himg, 1, BSZ, CONVB_N, BSZ, CONVB_NP, CONVB_N, stream);
    launch_gemm(himg, linW, h2, BSZ, LINNP, CONVB_NP, CONVB_NP, CONVB_NP, LINNP, 0, 0, 0, 1, stream);
    bias_rows_kernel<<<(BSZ * HID + 255) / 256, 256, 0, stream>>>(h2, lin_b, BSZ, HID, LINNP);
    launch_cvt(h2, h2bf, 1, BSZ, HID, BSZ, HIDP, LINNP, stream);

    // --- 20 heads: batched GEMM -> bias -> kwinners -> bf16
    launch_gemm(h2bf, hXw, hx, BSZ, HEADHP, HIDP, HIDP, HIDP, HEADHP,
                0, (long long)HEADHP * HIDP, (long long)BSZ * HEADHP, NHEADS, stream);
    bias_heads_kernel<<<(NHEADS * BSZ * HEADH + 255) / 256, 256, 0, stream>>>(hx, headX_b);
    kwinners_kernel<<<NHEADS * BSZ, 256, 0, stream>>>(hx, HEADHP, HEADH, 207, duty_x, 1, BSZ, HEADH);
    launch_cvt(hx, hxbf, 1, NHEADS * BSZ, HEADH, NHEADS * BSZ, HEADHP, HEADHP, stream);

    // --- final head output GEMM with fused bias -> d_out [20,256,16]
    dim3 gO(2, 1, NHEADS);
    gemm_headO_kernel<<<gO, 256, 0, stream>>>(hxbf, hOw, headO_b, out);
    (void)in_sizes; (void)n_in; (void)out_size;
}
